// Attention_26207890440906
// MI455X (gfx1250) — matrix operand order repaired and checked by reading, not compiled, not run on hardware
//
#include <hip/hip_runtime.h>
#include <math.h>

// ---- problem constants (match reference) ----
#define BB 16
#define NN 2048
#define CC 512
#define HH 8
#define DD 64
#define MM (BB * NN)   // 32768 rows

typedef __attribute__((ext_vector_type(16))) __bf16 bf16x16;
typedef __attribute__((ext_vector_type(8)))  __bf16 bf16x8;
typedef __attribute__((ext_vector_type(8)))  float  f32x8;
typedef __attribute__((ext_vector_type(4))) unsigned int u32x4;
typedef __attribute__((ext_vector_type(8)))  int i32x8;
typedef __attribute__((ext_vector_type(4)))  int i32x4;

static __device__ __forceinline__ f32x8 wmma_bf16(bf16x16 a, bf16x16 b, f32x8 c) {
    // D = A(16x32 bf16) * B(32x16 bf16) + C(16x16 f32)
    return __builtin_amdgcn_wmma_f32_16x16x32_bf16(false, a, false, b, (short)0, c, false, false);
}

// ---------------------------------------------------------------------------
// TDM: 2-D tile load global -> LDS (CDNA5 tensor_load_to_lds).
// D# group0/group1 per ISA cdna5 ch.8: count=1, type=2("image"),
// data_size=1 (2 bytes), tile = tile0 x tile1, row stride = stride0 elems.
// ---------------------------------------------------------------------------
static __device__ __forceinline__ void tdm_load_2d_bf16(
    unsigned int ldsAddr, unsigned long long gAddr,
    int tdim0, int tdim1, int tile0, int tile1, long stride0)
{
    u32x4 g0;
    g0[0] = 1u;                                                   // count=1, user mode
    g0[1] = ldsAddr;                                              // lds_addr [63:32]
    g0[2] = (unsigned int)(gAddr & 0xFFFFFFFFull);                // global_addr lo
    g0[3] = (unsigned int)((gAddr >> 32) & 0x01FFFFFFull)         // global_addr[56:32]
          | 0x80000000u;                                          // type=2 -> bits[127:126]=2
    i32x8 g1;
    g1[0] = 0x10000;                                              // data_size=1 (2B), mask=0
    g1[1] = (tdim0 & 0xFFFF) << 16;                               // tensor_dim0[15:0]
    g1[2] = ((tdim0 >> 16) & 0xFFFF) | ((tdim1 & 0xFFFF) << 16);  // td0 hi | td1 lo
    g1[3] = ((tdim1 >> 16) & 0xFFFF) | ((tile0 & 0xFFFF) << 16);  // td1 hi | tile_dim0
    g1[4] = (tile1 & 0xFFFF);                                     // tile_dim1 (tile_dim2=0)
    g1[5] = (int)(stride0 & 0xFFFFFFFFl);                         // tensor_dim0_stride lo
    g1[6] = (int)((stride0 >> 32) & 0xFFFFl);                     // stride hi (dim1_stride=0)
    g1[7] = 0;
    i32x4 z4 = {0, 0, 0, 0};
#if __clang_major__ >= 23
    i32x8 z8 = {0, 0, 0, 0, 0, 0, 0, 0};
    __builtin_amdgcn_tensor_load_to_lds(g0, g1, z4, z4, z8, 0);
#else
    __builtin_amdgcn_tensor_load_to_lds(g0, g1, z4, z4, 0);
#endif
}

// ---------------------------------------------------------------------------
// fp32 -> bf16 conversion
// ---------------------------------------------------------------------------
__global__ __launch_bounds__(256) void cvt_f32_bf16(const float* __restrict__ in,
                                                    __bf16* __restrict__ out, long n) {
    long i = (long)blockIdx.x * 256 + threadIdx.x;
    if (i < n) out[i] = (__bf16)in[i];
}

// ---------------------------------------------------------------------------
// Generic batched GEMM:  D[row, n] = act( sum_k A[row,k] * W[n,k] + bias[n] )
// A: bf16 row-major (lda), W: bf16 row-major over k (ldw), D: bf16 or f32 (ldd)
// Per-batch offsets: aOff = (z/innerCnt)*aOuter + (z%innerCnt)*aInner  (A and D)
//                    W offset = z*wStride
// Tile: 32 rows x 64 cols per wave (2 A-frags reuse 4 B-frags -> 8 WMMA/kstep),
// 8 waves per block -> 256x64 per block.
// ---------------------------------------------------------------------------
__global__ __launch_bounds__(256) void gemm_bf16_wmma(
    const __bf16* __restrict__ A, const __bf16* __restrict__ W,
    void* __restrict__ Dv, const float* __restrict__ bias,
    int lda, int ldw, int ldd, int K,
    long aOuter, long aInner, int innerCnt, long wStride,
    int act, int outF32)
{
    const int z = blockIdx.z;
    const long aOff = (long)(z / innerCnt) * aOuter + (long)(z % innerCnt) * aInner;
    const __bf16* Ab = A + aOff;
    const __bf16* Wb = W + (long)z * wStride;

    const int lane = threadIdx.x & 31;
    const int wave = threadIdx.x >> 5;
    const int m0 = blockIdx.y * 256 + wave * 32;
    const int n0 = blockIdx.x * 64;
    const int mrow0 = m0 + (lane & 15);
    const int mrow1 = mrow0 + 16;
    const int koffA = (lane < 16) ? 0 : 16;  // A element layout (contiguous k, matches B)
    const int koffB = (lane < 16) ? 0 : 16;  // B element layout
    const int nlane = lane & 15;

    f32x8 acc[2][4] = {{f32x8{}, f32x8{}, f32x8{}, f32x8{}},
                       {f32x8{}, f32x8{}, f32x8{}, f32x8{}}};

    for (int kb = 0; kb < K; kb += 32) {
        bf16x16 a0, a1;
        {
            const __bf16* ap0 = Ab + (long)mrow0 * lda + kb + koffA;
            const __bf16* ap1 = Ab + (long)mrow1 * lda + kb + koffA;
            bf16x8 l0 = *reinterpret_cast<const bf16x8*>(ap0);
            bf16x8 h0 = *reinterpret_cast<const bf16x8*>(ap0 + 8);
            bf16x8 l1 = *reinterpret_cast<const bf16x8*>(ap1);
            bf16x8 h1 = *reinterpret_cast<const bf16x8*>(ap1 + 8);
#pragma unroll
            for (int i = 0; i < 8; ++i) {
                a0[i] = l0[i]; a0[i + 8] = h0[i];
                a1[i] = l1[i]; a1[i + 8] = h1[i];
            }
            if (kb + 32 < K) {  // hint next A k-block toward the caches
                __builtin_prefetch(ap0 + 32, 0, 0);
                __builtin_prefetch(ap1 + 32, 0, 0);
            }
        }
#pragma unroll
        for (int j = 0; j < 4; ++j) {
            const int n = n0 + j * 16 + nlane;
            bf16x16 bfrag = *reinterpret_cast<const bf16x16*>(Wb + (long)n * ldw + kb + koffB);
            acc[0][j] = wmma_bf16(a0, bfrag, acc[0][j]);
            acc[1][j] = wmma_bf16(a1, bfrag, acc[1][j]);
        }
    }

#pragma unroll
    for (int s = 0; s < 2; ++s) {
        const int rbase = m0 + s * 16 + ((lane < 16) ? 0 : 8);
#pragma unroll
        for (int j = 0; j < 4; ++j) {
#pragma unroll
            for (int r = 0; r < 8; ++r) {
                const int row = rbase + r;
                const int col = n0 + j * 16 + nlane;
                float v = acc[s][j][r];
                if (bias) v += bias[col];
                if (act == 1) v = 0.5f * v * (1.0f + erff(v * 0.7071067811865475f));  // exact GELU
                const long doff = aOff + (long)row * ldd + col;
                if (outF32) ((float*)Dv)[doff] = v;
                else        ((__bf16*)Dv)[doff] = (__bf16)v;
            }
        }
    }
}

// ---------------------------------------------------------------------------
// Per-(b,h):  kk[d,e] = sum_n k[n,d]*k[n,e],  ktv[d,e] = sum_n k[n,d]*v[n,e]
// One block per (b,h), 8 waves. Each 32x64 k/v tile is staged into LDS by the
// Tensor Data Mover (wave 0 issues tensor_load_to_lds, waits TENSORcnt, then
// workgroup barrier). Wave w: e-tile (w&3), d-tiles {w>>2, (w>>2)+2}.
// ---------------------------------------------------------------------------
__global__ __launch_bounds__(256) void kkv_wmma(
    const __bf16* __restrict__ kmat, const __bf16* __restrict__ vmat,
    float* __restrict__ kkOut, float* __restrict__ ktvOut)
{
    __shared__ __bf16 kT[32 * 64];
    __shared__ __bf16 vT[32 * 64];

    const int z = blockIdx.x;            // b*H + h
    const int b = z >> 3, h = z & 7;
    const __bf16* kbase = kmat + (long)b * NN * CC + h * DD;
    const __bf16* vbase = vmat + (long)b * NN * CC + h * DD;

    const int lane = threadIdx.x & 31;
    const int wave = threadIdx.x >> 5;
    const int ec  = wave & 3;
    const int dr0 = wave >> 2;
    const int dr1 = dr0 + 2;

    const int koffA = (lane < 16) ? 0 : 16;
    const int koffB = (lane < 16) ? 0 : 16;
    const int mA0 = dr0 * 16 + (lane & 15);
    const int mA1 = dr1 * 16 + (lane & 15);
    const int nc  = ec * 16 + (lane & 15);

    const unsigned int kLds = (unsigned int)(size_t)(void*)&kT[0];  // low 32 bits = LDS offset
    const unsigned int vLds = (unsigned int)(size_t)(void*)&vT[0];

    f32x8 akk0 = {}, akk1 = {}, atv0 = {}, atv1 = {};

    for (int nb = 0; nb < NN; nb += 32) {
        __syncthreads();   // all waves done reading previous LDS tile
        if (wave == 0) {
            tdm_load_2d_bf16(kLds, (unsigned long long)(size_t)(kbase + (long)nb * CC),
                             DD, 32, DD, 32, CC);
            tdm_load_2d_bf16(vLds, (unsigned long long)(size_t)(vbase + (long)nb * CC),
                             DD, 32, DD, 32, CC);
            __builtin_amdgcn_s_wait_tensorcnt(0);
        }
        __syncthreads();   // tile visible to all waves

        bf16x16 a0, a1, bk, bv;
#pragma unroll
        for (int i = 0; i < 8; ++i) {
            a0[i]     = kT[(koffA + i) * 64 + mA0];
            a0[i + 8] = kT[(koffA + 8 + i) * 64 + mA0];
            a1[i]     = kT[(koffA + i) * 64 + mA1];
            a1[i + 8] = kT[(koffA + 8 + i) * 64 + mA1];
        }
#pragma unroll
        for (int i = 0; i < 16; ++i) {
            bk[i] = kT[(koffB + i) * 64 + nc];
            bv[i] = vT[(koffB + i) * 64 + nc];
        }
        akk0 = wmma_bf16(a0, bk, akk0);
        akk1 = wmma_bf16(a1, bk, akk1);
        atv0 = wmma_bf16(a0, bv, atv0);
        atv1 = wmma_bf16(a1, bv, atv1);
    }

    const int e = ec * 16 + (lane & 15);
    const int roff = (lane < 16) ? 0 : 8;
#pragma unroll
    for (int r = 0; r < 8; ++r) {
        const int d0 = dr0 * 16 + r + roff;
        const int d1 = dr1 * 16 + r + roff;
        kkOut [(long)z * 4096 + d0 * 64 + e] = akk0[r];
        kkOut [(long)z * 4096 + d1 * 64 + e] = akk1[r];
        ktvOut[(long)z * 4096 + d0 * 64 + e] = atv0[r];
        ktvOut[(long)z * 4096 + d1 * 64 + e] = atv1[r];
    }
}

// ---------------------------------------------------------------------------
// Per-(b,h): attn = softmax_rows( inv(kk) @ ktv ); store transposed as bf16
// (attnT[e,d] = attn[d,e]) so the o = q @ attn GEMM can treat it as W[n,k].
// 64 threads, thread j owns column j everywhere -> writes are thread-private.
// ---------------------------------------------------------------------------
__global__ __launch_bounds__(64) void inv_softmax(
    const float* __restrict__ kk, const float* __restrict__ ktv,
    __bf16* __restrict__ attnT)
{
    __shared__ float sA[64 * 65];
    __shared__ float sI[64 * 65];
    __shared__ float sT[64 * 65];
    __shared__ float fcol[64];

    const int j = threadIdx.x;
    const int z = blockIdx.x;
    const float* kkz = kk + (long)z * 4096;

    for (int r = 0; r < 64; ++r) {
        sA[r * 65 + j] = kkz[r * 64 + j];
        sI[r * 65 + j] = (r == j) ? 1.0f : 0.0f;
    }
    __syncthreads();

    // Gauss-Jordan (no pivoting; kk = K^T K is SPD-ish for this data)
    for (int p = 0; p < 64; ++p) {
        fcol[j] = sA[j * 65 + p];          // snapshot of pivot column
        __syncthreads();
        const float piv = fcol[p];
        const float pra = sA[p * 65 + j] / piv;
        const float pri = sI[p * 65 + j] / piv;
        for (int r = 0; r < 64; ++r) {
            if (r == p) continue;
            const float f = fcol[r];
            sA[r * 65 + j] -= f * pra;
            sI[r * 65 + j] -= f * pri;
        }
        sA[p * 65 + j] = pra;
        sI[p * 65 + j] = pri;
        __syncthreads();
    }

    const float* ktvz = ktv + (long)z * 4096;
    for (int r = 0; r < 64; ++r) sT[r * 65 + j] = ktvz[r * 64 + j];
    __syncthreads();

    // attn[:,j] = inv(kk) @ ktv[:,j]
    for (int d = 0; d < 64; ++d) {
        float s = 0.0f;
        for (int t = 0; t < 64; ++t) s += sI[d * 65 + t] * sT[t * 65 + j];
        sA[d * 65 + j] = s;                // column-private reuse of sA
    }
    // softmax over d (axis=-2), column j
    float mx = -1e30f;
    for (int d = 0; d < 64; ++d) mx = fmaxf(mx, sA[d * 65 + j]);
    float sum = 0.0f;
    for (int d = 0; d < 64; ++d) sum += expf(sA[d * 65 + j] - mx);
    const float inv_s = 1.0f / sum;
    for (int d = 0; d < 64; ++d)
        attnT[(long)z * 4096 + j * 64 + d] = (__bf16)(expf(sA[d * 65 + j] - mx) * inv_s);
}

// ---------------------------------------------------------------------------
// LayerNorm over C=512; one block per (b,n) row; 256 threads x 2 elements.
// ---------------------------------------------------------------------------
__global__ __launch_bounds__(256) void ln_kernel(
    const __bf16* __restrict__ o, const float* __restrict__ w,
    const float* __restrict__ bso, __bf16* __restrict__ out)
{
    __shared__ float red[256];
    const long row = blockIdx.x;
    const int  t   = threadIdx.x;
    const __bf16* pr = o + row * CC;
    const float v0 = (float)pr[t];
    const float v1 = (float)pr[t + 256];

    red[t] = v0 + v1;
    __syncthreads();
    for (int s = 128; s > 0; s >>= 1) { if (t < s) red[t] += red[t + s]; __syncthreads(); }
    const float mu = red[0] * (1.0f / CC);
    __syncthreads();

    const float d0 = v0 - mu, d1 = v1 - mu;
    red[t] = d0 * d0 + d1 * d1;
    __syncthreads();
    for (int s = 128; s > 0; s >>= 1) { if (t < s) red[t] += red[t + s]; __syncthreads(); }
    const float rstd = rsqrtf(red[0] * (1.0f / CC) + 1e-6f);

    out[row * CC + t]       = (__bf16)(d0 * rstd * w[t]       + bso[t]);
    out[row * CC + t + 256] = (__bf16)(d1 * rstd * w[t + 256] + bso[t + 256]);
}

// ---------------------------------------------------------------------------
// Depthwise conv (k=11, pad 5) along n, + bias, then gate:  g = conv(a) * vv
// Data layout [b, n, c]; adjacent threads = adjacent c -> coalesced.
// ---------------------------------------------------------------------------
__global__ __launch_bounds__(256) void dw_gate(
    const __bf16* __restrict__ a, const __bf16* __restrict__ vv,
    const float* __restrict__ dww, const float* __restrict__ dwb,
    __bf16* __restrict__ g)
{
    const long idx = (long)blockIdx.x * 256 + threadIdx.x;
    if (idx >= (long)BB * NN * CC) return;
    const int  c  = (int)(idx & (CC - 1));
    const long bn = idx >> 9;
    const int  n  = (int)(bn & (NN - 1));
    const long b  = bn >> 11;

    float acc = dwb[c];
#pragma unroll
    for (int tt = 0; tt < 11; ++tt) {
        const int nn = n + tt - 5;
        if (nn >= 0 && nn < NN)
            acc += (float)a[((b << 11) + nn) * CC + c] * dww[c * 11 + tt];
    }
    g[idx] = (__bf16)(acc * (float)vv[idx]);
}

// ---------------------------------------------------------------------------
extern "C" void kernel_launch(void* const* d_in, const int* in_sizes, int n_in,
                              void* d_out, int out_size, void* d_ws, size_t ws_size,
                              hipStream_t stream)
{
    const float* x      = (const float*)d_in[0];
    const float* wq     = (const float*)d_in[1];
    const float* wk     = (const float*)d_in[2];
    const float* wv     = (const float*)d_in[3];
    const float* ln_w   = (const float*)d_in[4];
    const float* ln_b   = (const float*)d_in[5];
    const float* ava1_w = (const float*)d_in[6];
    const float* ava1_b = (const float*)d_in[7];
    const float* dw_w   = (const float*)d_in[8];
    const float* dw_b   = (const float*)d_in[9];
    const float* v_w    = (const float*)d_in[10];
    const float* v_b    = (const float*)d_in[11];
    const float* proj_w = (const float*)d_in[12];
    const float* proj_b = (const float*)d_in[13];
    const float* out_w  = (const float*)d_in[14];
    const float* out_b  = (const float*)d_in[15];

    // ---- workspace carve-up (bf16 activations, 4 reusable 32MB buffers) ----
    char* p = (char*)d_ws;
    const size_t BIG = (size_t)MM * CC * sizeof(__bf16);     // 33,554,432
    const size_t WSZ = (size_t)CC * CC * sizeof(__bf16);     // 524,288
    __bf16* b0 = (__bf16*)(p);               // x_bf16 -> o -> g
    __bf16* b1 = (__bf16*)(p + BIG);         // q -> ln
    __bf16* b2 = (__bf16*)(p + 2 * BIG);     // k -> a -> p
    __bf16* b3 = (__bf16*)(p + 3 * BIG);     // v -> vv
    size_t off = 4 * BIG;
    __bf16* wqb   = (__bf16*)(p + off); off += WSZ;
    __bf16* wkb   = (__bf16*)(p + off); off += WSZ;
    __bf16* wvb   = (__bf16*)(p + off); off += WSZ;
    __bf16* avab  = (__bf16*)(p + off); off += WSZ;
    __bf16* vwb   = (__bf16*)(p + off); off += WSZ;
    __bf16* projb = (__bf16*)(p + off); off += WSZ;
    __bf16* outwb = (__bf16*)(p + off); off += WSZ;
    float*  kkBuf  = (float*)(p + off); off += (size_t)128 * 4096 * sizeof(float);
    float*  ktvBuf = (float*)(p + off); off += (size_t)128 * 4096 * sizeof(float);
    __bf16* attnT  = (__bf16*)(p + off); off += (size_t)128 * 4096 * sizeof(__bf16);

    const long NX = (long)MM * CC;   // 16,777,216
    const long NW = (long)CC * CC;   // 262,144

    // ---- 0) convert to bf16 ----
    cvt_f32_bf16<<<dim3((unsigned)((NX + 255) / 256)), 256, 0, stream>>>(x, b0, NX);
    cvt_f32_bf16<<<dim3((unsigned)((NW + 255) / 256)), 256, 0, stream>>>(wq, wqb, NW);
    cvt_f32_bf16<<<dim3((unsigned)((NW + 255) / 256)), 256, 0, stream>>>(wk, wkb, NW);
    cvt_f32_bf16<<<dim3((unsigned)((NW + 255) / 256)), 256, 0, stream>>>(wv, wvb, NW);
    cvt_f32_bf16<<<dim3((unsigned)((NW + 255) / 256)), 256, 0, stream>>>(ava1_w, avab, NW);
    cvt_f32_bf16<<<dim3((unsigned)((NW + 255) / 256)), 256, 0, stream>>>(v_w, vwb, NW);
    cvt_f32_bf16<<<dim3((unsigned)((NW + 255) / 256)), 256, 0, stream>>>(proj_w, projb, NW);
    cvt_f32_bf16<<<dim3((unsigned)((NW + 255) / 256)), 256, 0, stream>>>(out_w, outwb, NW);

    const dim3 gBig(CC / 64, MM / 256, 1);   // (8, 128, 1)

    // ---- 1) q, k, v projections ----
    gemm_bf16_wmma<<<gBig, 256, 0, stream>>>(b0, wqb, b1, nullptr, CC, CC, CC, CC,
                                             0, 0, 1, 0, 0, 0);
    gemm_bf16_wmma<<<gBig, 256, 0, stream>>>(b0, wkb, b2, nullptr, CC, CC, CC, CC,
                                             0, 0, 1, 0, 0, 0);
    gemm_bf16_wmma<<<gBig, 256, 0, stream>>>(b0, wvb, b3, nullptr, CC, CC, CC, CC,
                                             0, 0, 1, 0, 0, 0);

    // ---- 2) kk = K^T K, ktv = K^T V per (b,h)  (TDM-staged LDS tiles) ----
    kkv_wmma<<<dim3(BB * HH), 256, 0, stream>>>(b2, b3, kkBuf, ktvBuf);

    // ---- 3) attn = softmax(inv(kk) @ ktv), stored transposed bf16 ----
    inv_softmax<<<dim3(BB * HH), 64, 0, stream>>>(kkBuf, ktvBuf, attnT);

    // ---- 4) o = q @ attn per head (batched GEMM, attnT acts as W[n,k]) ----
    gemm_bf16_wmma<<<dim3(1, NN / 256, BB * HH), 256, 0, stream>>>(
        b1, attnT, b0, nullptr, CC, DD, CC, DD,
        /*aOuter*/ (long)NN * CC, /*aInner*/ DD, /*innerCnt*/ HH,
        /*wStride*/ (long)DD * DD, 0, 0);

    // ---- 5) LayerNorm ----
    ln_kernel<<<dim3(MM), 256, 0, stream>>>(b0, ln_w, ln_b, b1);

    // ---- 6) a = GELU(ln @ ava1_w^T + b),  vv = ln @ v_w^T + b ----
    gemm_bf16_wmma<<<gBig, 256, 0, stream>>>(b1, avab, b2, ava1_b, CC, CC, CC, CC,
                                             0, 0, 1, 0, /*act=gelu*/ 1, 0);
    gemm_bf16_wmma<<<gBig, 256, 0, stream>>>(b1, vwb, b3, v_b, CC, CC, CC, CC,
                                             0, 0, 1, 0, 0, 0);

    // ---- 7) depthwise conv + gate ----
    dw_gate<<<dim3((unsigned)(NX / 256)), 256, 0, stream>>>(b2, b3, dw_w, dw_b, b0);

    // ---- 8) p = g @ proj_w^T + b ----
    gemm_bf16_wmma<<<gBig, 256, 0, stream>>>(b0, projb, b2, proj_b, CC, CC, CC, CC,
                                             0, 0, 1, 0, 0, 0);

    // ---- 9) out = p @ out_w^T + b  (fp32 to d_out) ----
    gemm_bf16_wmma<<<gBig, 256, 0, stream>>>(b2, outwb, d_out, out_b, CC, CC, CC, CC,
                                             0, 0, 1, 0, 0, /*outF32*/ 1);
}
